// SelfAttention_40956808135123
// MI455X (gfx1250) — matrix operand order repaired and checked by reading, not compiled, not run on hardware
//
#include <hip/hip_runtime.h>
#include <hip/hip_bf16.h>

typedef __bf16 bf16_t;
typedef __attribute__((ext_vector_type(16))) __bf16 v16bf;
typedef __attribute__((ext_vector_type(8)))  float  v8f;

#define DEV __device__ __forceinline__

constexpr int BATCH  = 4;
constexpr int SEQ    = 2048;
constexpr int DMODEL = 1024;
constexpr int NH     = 16;
constexpr int HD     = 64;
constexpr int TLD    = 72;            // padded stride (halves) for K-major tiles; 144 B, 16 B aligned
constexpr float SCALE_INV = 0.125f;   // 1/sqrt(64)

#if defined(__has_builtin)
#if __has_builtin(__builtin_amdgcn_tensor_load_to_lds) && \
    __has_builtin(__builtin_amdgcn_s_wait_tensorcnt)
#define USE_TDM 1
#endif
#endif

DEV int lane_id() { return threadIdx.x & 31; }
DEV int wave_id() { return threadIdx.x >> 5; }

union Frag256 { uint4 u[2]; v16bf v; };
union Pack4   { bf16_t h[4]; uint2 u; };

// ---------------------------------------------------------------------------
// WMMA fragment helpers (CDNA5 wave32 layouts, CDNA5 ISA §7.12.2)
// ---------------------------------------------------------------------------

// A-matrix 16x32 bf16, source row-major [row][k]:
// lane L holds row L%16; element i -> k = i + 8*(L>=16) + 8*(i>=8)
DEV v16bf load_A_frag(const bf16_t* smem, int row0, int k0, int ld) {
  const int l  = lane_id();
  const int r  = l & 15;
  const int hs = (l >> 4) << 3;           // 0 or 8
  const bf16_t* p = smem + (row0 + r) * ld + k0 + hs;
  Frag256 f;
  f.u[0] = *(const uint4*)p;              // k = hs .. hs+7
  f.u[1] = *(const uint4*)(p + 16);       // k = hs+16 .. hs+23
  return f.v;
}

// B-matrix 32x16 bf16, source K-major [k][n] with stride ld:
// lane L holds column L%16; element i -> k = i + 8*(L>=16) + 8*(i>=8)
DEV v16bf load_B_frag(const bf16_t* smem, int k0, int n0, int ld) {
  const int l  = lane_id();
  const int hs = (l >> 4) << 3;
  const bf16_t* p = smem + (k0 + hs) * ld + n0 + (l & 15);
  v16bf v;
#pragma unroll
  for (int i = 0; i < 8; ++i) {
    v[i]     = p[i * ld];
    v[i + 8] = p[(i + 16) * ld];
  }
  return v;
}

DEV v8f wmma_bf16(v16bf a, v16bf b, v8f c) {
  return __builtin_amdgcn_wmma_f32_16x16x32_bf16(false, a, false, b, (short)0, c,
                                                 false, false);
}

// ---------------------------------------------------------------------------
// Tensor Data Mover (CDNA5 ISA): build D# and issue
// TENSOR_LOAD_TO_LDS.  Issued by one wave; completion via s_wait_tensorcnt.
// clang-23 toolchain: 6-arg builtin (g0 u32x4, g1 i32x8, g2 i32x4, g3 i32x4,
// extra i32x8, i32 cpol).
// ---------------------------------------------------------------------------
#ifdef USE_TDM
typedef unsigned int v4u  __attribute__((ext_vector_type(4)));
typedef int          v8i_ __attribute__((ext_vector_type(8)));
typedef int          v4i_ __attribute__((ext_vector_type(4)));

DEV void tdm_load(const void* gsrc, void* lds_dst,
                  unsigned dataSizeCode,                  // 0:1B 1:2B 2:4B 3:8B
                  unsigned tensorDim0, unsigned tensorDim1,
                  unsigned tileDim0, unsigned tileDim1,
                  unsigned long long stride0,             // dim0 stride (elements)
                  unsigned padEnable, unsigned padInterval, unsigned padAmount) {
  const unsigned long long ga = (unsigned long long)(size_t)gsrc;
  const unsigned ldsa = (unsigned)(size_t)lds_dst;        // truncate -> LDS offset

  v4u g0;
  g0[0] = 1u;                                             // count=1 (valid user D#)
  g0[1] = ldsa;                                           // lds_addr [63:32]
  g0[2] = (unsigned)(ga & 0xffffffffu);                   // global_addr[31:0]
  g0[3] = (unsigned)((ga >> 32) & 0x1ffffffu) | (2u << 30); // addr[56:32] | type=2

  const unsigned w0 = (dataSizeCode << 16) | (padEnable << 20) |
                      (padInterval << 22) | (padAmount << 25);
  const unsigned w1 = (tensorDim0 & 0xffffu) << 16;       // tensor_dim0[15:0] @63:48
  const unsigned w2 = (tensorDim0 >> 16) | ((tensorDim1 & 0xffffu) << 16);
  const unsigned w3 = (tensorDim1 >> 16) | (tileDim0 << 16);
  const unsigned w4 = tileDim1;                           // tile_dim1 | tile_dim2=0
  const unsigned w5 = (unsigned)(stride0 & 0xffffffffu);  // dim0_stride[31:0]
  const unsigned w6 = (unsigned)(stride0 >> 32);          // dim0_stride[47:32]
  v8i_ g1;
  g1[0] = (int)w0; g1[1] = (int)w1; g1[2] = (int)w2; g1[3] = (int)w3;
  g1[4] = (int)w4; g1[5] = (int)w5; g1[6] = (int)w6; g1[7] = 0;
  const v4i_ z4 = {0, 0, 0, 0};
  const v8i_ z8 = {0, 0, 0, 0, 0, 0, 0, 0};
  __builtin_amdgcn_tensor_load_to_lds(g0, g1, z4, z4, z8, 0);
}
#endif

// ---------------------------------------------------------------------------
// Tiled GEMM: C[m][n] = sum_k A[m][k] * W[n][k] + bias[n]
// MODE 0: A fp32, out bf16 scaled, layout [B,H,S,Hd]   (Q and V projections)
// MODE 2: A fp32, out bf16,        layout [B,H,Hd,S]   (K projection, transposed)
// MODE 1: A bf16, out fp32 row-major [M][N]            (final projection)
// Block: 128 threads (4 waves), 64x64 C tile, K chunks of 64.
// ---------------------------------------------------------------------------
template <int MODE>
__global__ __launch_bounds__(128)
void gemm64(const void* __restrict__ Asrc, const float* __restrict__ W,
            const float* __restrict__ bias, void* __restrict__ dst,
            float outScale) {
  constexpr int K = DMODEL;
  __shared__ bf16_t sA[64 * 64];        // row-major [m][k]
  __shared__ bf16_t sBt[64 * TLD];      // K-major   [k][n]

  const int n0 = blockIdx.x * 64;
  const int m0 = blockIdx.y * 64;
  const int w  = wave_id();
  const int l  = lane_id();

  v8f acc[4];
#pragma unroll
  for (int i = 0; i < 4; ++i) acc[i] = {};

  for (int kk = 0; kk < K; kk += 64) {
    __syncthreads();
    // ---- A tile 64x64 -> sA (row-major) ----
    if (MODE != 1) {
      const float* A = (const float*)Asrc;
      for (int j = threadIdx.x; j < 1024; j += 128) {   // 1024 float4s
        const int r  = j >> 4;
        const int c0 = (j & 15) << 2;
        const float4 f = *(const float4*)(A + (size_t)(m0 + r) * K + kk + c0);
        Pack4 p;
        p.h[0] = (bf16_t)f.x; p.h[1] = (bf16_t)f.y;
        p.h[2] = (bf16_t)f.z; p.h[3] = (bf16_t)f.w;
        *(uint2*)(sA + r * 64 + c0) = p.u;
      }
    } else {
      const bf16_t* A = (const bf16_t*)Asrc;
      for (int j = threadIdx.x; j < 512; j += 128) {    // 512 uint4s
        const int r  = j >> 3;
        const int c0 = (j & 7) << 3;
        ((uint4*)sA)[j] = *(const uint4*)(A + (size_t)(m0 + r) * K + kk + c0);
      }
    }
    // ---- W tile 64(n) x 64(k) -> sBt transposed [k][n] ----
    for (int j = threadIdx.x; j < 1024; j += 128) {     // 1024 float4s
      const int n  = j >> 4;
      const int c0 = (j & 15) << 2;
      const float4 f = *(const float4*)(W + (size_t)(n0 + n) * K + kk + c0);
      sBt[(c0 + 0) * TLD + n] = (bf16_t)f.x;
      sBt[(c0 + 1) * TLD + n] = (bf16_t)f.y;
      sBt[(c0 + 2) * TLD + n] = (bf16_t)f.z;
      sBt[(c0 + 3) * TLD + n] = (bf16_t)f.w;
    }
    if (kk + 64 < K)
      __builtin_prefetch(W + (size_t)n0 * K + kk + 64, 0, 0);
    __syncthreads();

    const v16bf a0 = load_A_frag(sA, w * 16, 0, 64);
    const v16bf a1 = load_A_frag(sA, w * 16, 32, 64);
#pragma unroll
    for (int nf = 0; nf < 4; ++nf) {
      acc[nf] = wmma_bf16(a0, load_B_frag(sBt, 0,  nf * 16, TLD), acc[nf]);
      acc[nf] = wmma_bf16(a1, load_B_frag(sBt, 32, nf * 16, TLD), acc[nf]);
    }
  }

  // Epilogue. D layout: lane holds column l&15, rows (l>>4)*8 + r.
  const int ncol = l & 15;
  const int mo   = (l >> 4) * 8;
#pragma unroll
  for (int nf = 0; nf < 4; ++nf) {
    const int gn   = n0 + nf * 16 + ncol;
    const float bv = bias[gn];
#pragma unroll
    for (int r = 0; r < 8; ++r) {
      const int gm  = m0 + w * 16 + mo + r;
      const float v = (acc[nf][r] + bv) * outScale;
      if (MODE == 0) {
        const int b = gm >> 11, s = gm & (SEQ - 1);
        const int h = gn >> 6,  hd = gn & (HD - 1);
        ((bf16_t*)dst)[(((size_t)b * NH + h) * SEQ + s) * HD + hd] = (bf16_t)v;
      } else if (MODE == 2) {
        const int b = gm >> 11, s = gm & (SEQ - 1);
        const int h = gn >> 6,  hd = gn & (HD - 1);
        ((bf16_t*)dst)[(((size_t)b * NH + h) * HD + hd) * SEQ + s] = (bf16_t)v;
      } else {
        ((float*)dst)[(size_t)gm * DMODEL + gn] = v;
      }
    }
  }
}

// ---------------------------------------------------------------------------
// Fused attention: per (b,h) and 64-row query block.
// K is stored [B,H,Hd,S] in global; its LDS tile arrives K-major (padded to
// TLD) either via TDM 2D descriptor with LDS padding, or a strided copy.
// ---------------------------------------------------------------------------
__global__ __launch_bounds__(128)
void attention_kernel(const bf16_t* __restrict__ Qg, const bf16_t* __restrict__ Ktg,
                      const bf16_t* __restrict__ Vg, float* __restrict__ attn,
                      bf16_t* __restrict__ Wout) {
  __shared__ bf16_t sQ[64 * 64];        // row-major [q][hd]
  __shared__ bf16_t sKt[64 * TLD];      // K-major   [hd][key]
  __shared__ bf16_t sV[64 * 64];        // row-major [key][hd]
  __shared__ bf16_t sP[64 * 64];        // row-major [q][key]

  const int q0 = blockIdx.x * 64;
  const int bh = blockIdx.y;            // b*NH + h
  const int w  = wave_id();
  const int l  = lane_id();

  const bf16_t* Qbase  = Qg  + ((size_t)bh * SEQ + q0) * HD;
  const bf16_t* Ktbase = Ktg + (size_t)bh * HD * SEQ;   // rows: hd, stride SEQ
  const bf16_t* Vbase  = Vg  + (size_t)bh * SEQ * HD;

  // stage Q tile (contiguous 8 KB)
#ifdef USE_TDM
  if (threadIdx.x < 32) {
    tdm_load(Qbase, sQ, 3u, 1024u, 0u, 1024u, 0u, 1024ull, 0u, 0u, 0u);
    __builtin_amdgcn_s_wait_tensorcnt(0);
  }
#else
  {
    const uint4* s = (const uint4*)Qbase;
    uint4* d = (uint4*)sQ;
    for (int i = threadIdx.x; i < 512; i += 128) d[i] = s[i];
  }
#endif
  __syncthreads();
  const v16bf aq0 = load_A_frag(sQ, w * 16, 0, 64);
  const v16bf aq1 = load_A_frag(sQ, w * 16, 32, 64);

  float rM[8], rL[8];
#pragma unroll
  for (int r = 0; r < 8; ++r) { rM[r] = -3.0e38f; rL[r] = 0.f; }

  const int nTiles = SEQ / 64;

  // ---- pass 1: streaming softmax stats ----
  for (int t = 0; t < nTiles; ++t) {
    __syncthreads();
#ifdef USE_TDM
    if (threadIdx.x < 32) {
      // 2D tile: 64 rows (hd) x 64 keys, row stride SEQ elements; LDS pad:
      // 4 dwords after every 32 dwords -> row stride TLD halves.
      tdm_load(Ktbase + t * 64, sKt, 1u, (unsigned)SEQ, (unsigned)HD,
               64u, 64u, (unsigned long long)SEQ, 1u, 4u, 3u);
      __builtin_amdgcn_s_wait_tensorcnt(0);
    }
#else
    for (int j = threadIdx.x; j < 512; j += 128) {
      const int r  = j >> 3;
      const int c0 = (j & 7) << 3;
      *(uint4*)(sKt + r * TLD + c0) =
          *(const uint4*)(Ktbase + (size_t)r * SEQ + t * 64 + c0);
    }
    if (t + 1 < nTiles)
      __builtin_prefetch(Ktbase + (t + 1) * 64, 0, 0);
#endif
    __syncthreads();

    v8f dfr[4];
#pragma unroll
    for (int nf = 0; nf < 4; ++nf) {
      v8f c = {};
      c = wmma_bf16(aq0, load_B_frag(sKt, 0,  nf * 16, TLD), c);
      c = wmma_bf16(aq1, load_B_frag(sKt, 32, nf * 16, TLD), c);
      dfr[nf] = c;
    }
#pragma unroll
    for (int r = 0; r < 8; ++r) {
      float m = dfr[0][r];
#pragma unroll
      for (int nf = 1; nf < 4; ++nf) m = fmaxf(m, dfr[nf][r]);
#pragma unroll
      for (int x = 1; x < 16; x <<= 1) m = fmaxf(m, __shfl_xor(m, x));
      const float nM = fmaxf(rM[r], m);
      float p = 0.f;
#pragma unroll
      for (int nf = 0; nf < 4; ++nf) p += __expf(dfr[nf][r] - nM);
#pragma unroll
      for (int x = 1; x < 16; x <<= 1) p += __shfl_xor(p, x);
      rL[r] = rL[r] * __expf(rM[r] - nM) + p;
      rM[r] = nM;
    }
  }
  float invL[8];
#pragma unroll
  for (int r = 0; r < 8; ++r) invL[r] = 1.0f / rL[r];

  v8f acc[4];
#pragma unroll
  for (int i = 0; i < 4; ++i) acc[i] = {};

  const int ncol = l & 15;
  const int mo   = (l >> 4) * 8;

  // ---- pass 2: probs + attention output + A*V ----
  for (int t = 0; t < nTiles; ++t) {
    __syncthreads();
#ifdef USE_TDM
    if (threadIdx.x < 32) {
      tdm_load(Ktbase + t * 64, sKt, 1u, (unsigned)SEQ, (unsigned)HD,
               64u, 64u, (unsigned long long)SEQ, 1u, 4u, 3u);
      tdm_load(Vbase + (size_t)t * 64 * HD, sV, 3u, 1024u, 0u,
               1024u, 0u, 1024ull, 0u, 0u, 0u);
      __builtin_amdgcn_s_wait_tensorcnt(0);
    }
#else
    for (int j = threadIdx.x; j < 512; j += 128) {
      const int r  = j >> 3;
      const int c0 = (j & 7) << 3;
      *(uint4*)(sKt + r * TLD + c0) =
          *(const uint4*)(Ktbase + (size_t)r * SEQ + t * 64 + c0);
    }
    {
      const uint4* sv = (const uint4*)(Vbase + (size_t)t * 64 * HD);
      uint4* dv = (uint4*)sV;
      for (int i = threadIdx.x; i < 512; i += 128) dv[i] = sv[i];
    }
    if (t + 1 < nTiles) {
      __builtin_prefetch(Ktbase + (t + 1) * 64, 0, 0);
      __builtin_prefetch(Vbase + (size_t)(t + 1) * 64 * HD, 0, 0);
    }
#endif
    __syncthreads();

    v8f dfr[4];
#pragma unroll
    for (int nf = 0; nf < 4; ++nf) {
      v8f c = {};
      c = wmma_bf16(aq0, load_B_frag(sKt, 0,  nf * 16, TLD), c);
      c = wmma_bf16(aq1, load_B_frag(sKt, 32, nf * 16, TLD), c);
      dfr[nf] = c;
    }
#pragma unroll
    for (int nf = 0; nf < 4; ++nf) {
#pragma unroll
      for (int r = 0; r < 8; ++r) {
        const float p = __expf(dfr[nf][r] - rM[r]) * invL[r];
        const int qr = w * 16 + mo + r;
        const int kc = nf * 16 + ncol;
        attn[((size_t)bh * SEQ + (q0 + qr)) * SEQ + (size_t)t * 64 + kc] = p;
        sP[qr * 64 + kc] = (bf16_t)p;
      }
    }
    __syncthreads();

    const v16bf ap0 = load_A_frag(sP, w * 16, 0, 64);
    const v16bf ap1 = load_A_frag(sP, w * 16, 32, 64);
#pragma unroll
    for (int nf = 0; nf < 4; ++nf) {
      acc[nf] = wmma_bf16(ap0, load_B_frag(sV, 0,  nf * 16, 64), acc[nf]);
      acc[nf] = wmma_bf16(ap1, load_B_frag(sV, 32, nf * 16, 64), acc[nf]);
    }
  }

  // weighted -> [B, S, D] bf16 (column block h*HD)
  const int b = bh >> 4;
  const int h = bh & 15;
#pragma unroll
  for (int nf = 0; nf < 4; ++nf) {
#pragma unroll
    for (int r = 0; r < 8; ++r) {
      const int qr = q0 + w * 16 + mo + r;
      const int hd = nf * 16 + ncol;
      Wout[((size_t)b * SEQ + qr) * DMODEL + h * HD + hd] = (bf16_t)acc[nf][r];
    }
  }
}

// ---------------------------------------------------------------------------
extern "C" void kernel_launch(void* const* d_in, const int* in_sizes, int n_in,
                              void* d_out, int out_size, void* d_ws, size_t ws_size,
                              hipStream_t stream) {
  (void)in_sizes; (void)n_in; (void)out_size; (void)ws_size;
  const float* query = (const float*)d_in[0];
  const float* key   = (const float*)d_in[1];
  const float* value = (const float*)d_in[2];
  const float* wq_w  = (const float*)d_in[3];
  const float* wq_b  = (const float*)d_in[4];
  const float* wk_w  = (const float*)d_in[5];
  const float* wk_b  = (const float*)d_in[6];
  const float* wv_w  = (const float*)d_in[7];
  const float* wv_b  = (const float*)d_in[8];
  const float* fc_w  = (const float*)d_in[9];
  const float* fc_b  = (const float*)d_in[10];

  float* out  = (float*)d_out;
  float* attn = out + (size_t)BATCH * SEQ * DMODEL;

  const size_t qkvElems = (size_t)BATCH * SEQ * DMODEL;
  bf16_t* Qb  = (bf16_t*)d_ws;
  bf16_t* Ktb = Qb + qkvElems;          // [B,H,Hd,S]
  bf16_t* Vb  = Ktb + qkvElems;
  bf16_t* Wt  = Vb + qkvElems;

  const dim3 gGemm(DMODEL / 64, (BATCH * SEQ) / 64);
  const dim3 b128(128);

  gemm64<0><<<gGemm, b128, 0, stream>>>(query, wq_w, wq_b, Qb,  SCALE_INV);
  gemm64<2><<<gGemm, b128, 0, stream>>>(key,   wk_w, wk_b, Ktb, 1.0f);
  gemm64<0><<<gGemm, b128, 0, stream>>>(value, wv_w, wv_b, Vb,  1.0f);

  const dim3 gAtt(SEQ / 64, BATCH * NH);
  attention_kernel<<<gAtt, b128, 0, stream>>>(Qb, Ktb, Vb, attn, Wt);

  gemm64<1><<<gGemm, b128, 0, stream>>>(Wt, fc_w, fc_b, out, 1.0f);
}
